// NeuralODESolver_14731737825354
// MI455X (gfx1250) — matrix operand order repaired and checked by reading, not compiled, not run on hardware
//
#include <hip/hip_runtime.h>

// ---------------- CDNA5 WMMA types ----------------
typedef __attribute__((ext_vector_type(16))) _Float16 v16h;
typedef __attribute__((ext_vector_type(8)))  _Float16 v8h;
typedef __attribute__((ext_vector_type(8)))  float    v8f;

// ---------------- Tsit5 tableau / step -------------
#define DT_H (60.0f / 3600.0f)
#define CA21 0.161f
#define CA31 -0.008480655492356989f
#define CA32 0.335480655492357f
#define CA41 2.8971530571054935f
#define CA42 -6.359448489975075f
#define CA43 4.3622954328695815f
#define CA51 5.325864828439257f
#define CA52 -11.748883564062828f
#define CA53 7.4955393428898365f
#define CA54 -0.09249506636175525f
#define CA61 5.86145544294642f
#define CA62 -12.92096931784711f
#define CA63 8.159367898576159f
#define CA64 -0.071584973281401f
#define CA65 -0.028269050394068383f
#define CB1 0.09646076681806523f
#define CB2 0.01f
#define CB3 0.4798896504144996f
#define CB4 1.379008574103742f
#define CB5 -3.290069515436081f
#define CB6 2.324710524099774f

// Problem dims (fixed by reference)
#define DIMD 64
#define DIMU 64
#define DIMH 256
#define KIN  128   // D + U

// LDS: one reusable 64KB staging buffer + activation buffers
#define STAGE_ELEMS (32768)        // 64 KB as f16 (holds W1, half of W2, or W3)
#define SIN_ELEMS   (16 * KIN)     // 2048
#define SH_ELEMS    (16 * DIMH)    // 4096

static constexpr size_t LDS_BYTES =
    (size_t)(STAGE_ELEMS + SIN_ELEMS + SH_ELEMS + SH_ELEMS) * 2;  // 86016 B (~84 KB)
// Steady-state LDS is small: leaves room for a second resident workgroup per WGP
// (320 KB LDS) to hide barrier latency, instead of pinning 252 KB of dead staging.

__device__ __forceinline__ v8f wmma_f16(v16h a, v16h b, v8f c) {
  return __builtin_amdgcn_wmma_f32_16x16x32_f16(
      /*neg_a=*/false, a, /*neg_b=*/false, b,
      /*c_mod=*/(short)0, c, /*reuse_a=*/false, /*reuse_b=*/false);
}

// A-fragment (16-bit A 16x32): lane L holds row M=L&15; K = kc*32 + 16*(L>>4) + 0..15.
// Two adjacent 16B runs from a row-major [16][ldk] buffer.
__device__ __forceinline__ v16h afrag(const _Float16* base, int row, int ldk,
                                      int kc, int lhalf) {
  const _Float16* p = base + row * ldk + kc * 32 + 16 * lhalf;
  v8h lo = *(const v8h*)p;
  v8h hi = *(const v8h*)(p + 8);
  v16h a;
#pragma unroll
  for (int i = 0; i < 8; ++i) { a[i] = lo[i]; a[i + 8] = hi[i]; }
  return a;
}

// B-fragment (16-bit B 32x16): lane L holds column N=L&15, K = kc*32 + 16*(L>>4) + 0..15.
// One contiguous 32B run from row n of a row-major [N][ldk] weight (B[k][n] = W[n][k]).
__device__ __forceinline__ v16h bfrag(const _Float16* w, int ldk, int ncol,
                                      int kc, int lhalf) {
  return *(const v16h*)(w + ncol * ldk + kc * 32 + 16 * lhalf);
}

__device__ __forceinline__ void axpy(v8f& a, float c, const v8f& v) {
#pragma unroll
  for (int r = 0; r < 8; ++r) a[r] += c * v[r];
}

__global__ __launch_bounds__(256, 1)
void tsit5_neural_ode_kernel(const float* __restrict__ x0,
                             const float* __restrict__ u,
                             const float* __restrict__ W1, const float* __restrict__ b1,
                             const float* __restrict__ W2, const float* __restrict__ b2,
                             const float* __restrict__ W3, const float* __restrict__ b3,
                             float* __restrict__ out,
                             const int* __restrict__ num_steps) {
  extern __shared__ __align__(32) char smem[];
  _Float16* sStage = (_Float16*)smem;            // [32768] reusable weight staging
  _Float16* sIn = sStage + STAGE_ELEMS;          // [16][128] concat(y,u)
  _Float16* sH1 = sIn + SIN_ELEMS;               // [16][256]
  _Float16* sH2 = sH1 + SH_ELEMS;                // [16][256]

  const int tid   = threadIdx.x;
  const int lane  = tid & 31;
  const int wave  = tid >> 5;       // 8 waves
  const int lmod  = lane & 15;
  const int lhalf = lane >> 4;
  const int rowBase = blockIdx.x * 16;
  const bool owner  = wave < 4;     // waves 0..3 own the 4 N-tiles of D=64

  const int nCol  = wave * 32 + lmod;   // this wave's first L1/L2 output column (per lane)

  // ---- biases: direct per-lane global loads (init-only, tiny, L2-served) ----
  float bv1[2], bv2[2], bv3 = 0.f;
#pragma unroll
  for (int nt = 0; nt < 2; ++nt) {
    bv1[nt] = b1[nCol + nt * 16];
    bv2[nt] = b2[nCol + nt * 16];
  }
  if (owner) bv3 = b3[wave * 16 + lmod];

  // ---- u tile -> f16 LDS (persists across the whole integration) ----
  for (int i = tid; i < 16 * DIMU; i += 256) {
    int m = i >> 6, c = i & 63;
    sIn[m * KIN + DIMU + c] = (_Float16)u[(rowBase + m) * DIMU + c];
  }

  // ---- stage weights through the 64KB buffer, extracting this wave's B-fragments
  //      into registers (weights are loop-invariant across all 1536 f-evals; CDNA5's
  //      1024-VGPR/wave budget keeps them resident) ----
  v16h w1b[2][4], w2b[2][8], w3b[8] = {};

  // W1: [256][128] fits whole
  for (int i = tid; i < DIMH * KIN; i += 256) sStage[i] = (_Float16)W1[i];
  __syncthreads();
#pragma unroll
  for (int nt = 0; nt < 2; ++nt)
#pragma unroll
    for (int kc = 0; kc < 4; ++kc)
      w1b[nt][kc] = bfrag(sStage, KIN, nCol + nt * 16, kc, lhalf);

  // u part of layer-1 A never changes: hoist its two A-fragments now (sIn u-part is ready)
  v16h a_u0 = afrag(sIn, lmod, KIN, 2, lhalf);
  v16h a_u1 = afrag(sIn, lmod, KIN, 3, lhalf);
  __syncthreads();

  // W2: [256][256] in two 128-column halves
#pragma unroll 1
  for (int half = 0; half < 2; ++half) {
    for (int i = tid; i < DIMH * 128; i += 256) {
      int n = i >> 7, k = i & 127;
      sStage[i] = (_Float16)W2[n * DIMH + half * 128 + k];
    }
    __syncthreads();
#pragma unroll
    for (int nt = 0; nt < 2; ++nt)
#pragma unroll
      for (int kc = 0; kc < 4; ++kc)
        w2b[nt][half * 4 + kc] = bfrag(sStage, 128, nCol + nt * 16, kc, lhalf);
    __syncthreads();
  }

  // W3: [64][256] fits whole
  for (int i = tid; i < DIMD * DIMH; i += 256) sStage[i] = (_Float16)W3[i];
  __syncthreads();
  if (owner) {
#pragma unroll
    for (int kc = 0; kc < 8; ++kc)
      w3b[kc] = bfrag(sStage, DIMH, wave * 16 + lmod, kc, lhalf);
  }

  // ---- initial state in WMMA C-fragment layout: elem r <-> (M=8*lhalf+r, N=wave*16+lmod)
  v8f y = {};
  if (owner) {
    const int col = wave * 16 + lmod;
#pragma unroll
    for (int r = 0; r < 8; ++r)
      y[r] = x0[(rowBase + 8 * lhalf + r) * DIMD + col];
  }
  __syncthreads();

  // owners write stage input y_s (f32->f16) into the x-part of the concat buffer
  auto stage_in = [&](const v8f& ys) {
    const int col = wave * 16 + lmod;
#pragma unroll
    for (int r = 0; r < 8; ++r)
      sIn[(8 * lhalf + r) * KIN + col] = (_Float16)ys[r];
  };

  // relu + f32->f16 + store one 16x16 C-tile into a [16][DIMH] LDS buffer
  auto relu_store = [&](const v8f& acc, float bv, _Float16* dst, int n0) {
#pragma unroll
    for (int r = 0; r < 8; ++r) {
      float v = acc[r] + bv;
      v = v > 0.f ? v : 0.f;
      dst[(8 * lhalf + r) * DIMH + n0 + lmod] = (_Float16)v;
    }
  };

  // one MLP evaluation: k = W3*relu(W2*relu(W1*[y_s;u]+b1)+b2)+b3
  auto evalF = [&](v8f& kout) {
    __syncthreads();  // sIn ready
    {   // layer 1: K=128; y-part A-frags fresh, u-part A-frags hoisted
      v16h a[4];
      a[0] = afrag(sIn, lmod, KIN, 0, lhalf);
      a[1] = afrag(sIn, lmod, KIN, 1, lhalf);
      a[2] = a_u0;
      a[3] = a_u1;
      v8f acc0 = {}, acc1 = {};
#pragma unroll
      for (int kc = 0; kc < 4; ++kc) {
        acc0 = wmma_f16(a[kc], w1b[0][kc], acc0);
        acc1 = wmma_f16(a[kc], w1b[1][kc], acc1);
      }
      relu_store(acc0, bv1[0], sH1, wave * 32);
      relu_store(acc1, bv1[1], sH1, wave * 32 + 16);
    }
    __syncthreads();
    {   // layer 2: K=256, two interleaved N-tile accumulator chains
      v16h a[8];
#pragma unroll
      for (int kc = 0; kc < 8; ++kc) a[kc] = afrag(sH1, lmod, DIMH, kc, lhalf);
      v8f acc0 = {}, acc1 = {};
#pragma unroll
      for (int kc = 0; kc < 8; ++kc) {
        acc0 = wmma_f16(a[kc], w2b[0][kc], acc0);
        acc1 = wmma_f16(a[kc], w2b[1][kc], acc1);
      }
      relu_store(acc0, bv2[0], sH2, wave * 32);
      relu_store(acc1, bv2[1], sH2, wave * 32 + 16);
    }
    __syncthreads();
    if (owner) {  // layer 3: K=256, waves 0..3 each one N-tile, result stays in VGPRs
      v16h a[8];
#pragma unroll
      for (int kc = 0; kc < 8; ++kc) a[kc] = afrag(sH2, lmod, DIMH, kc, lhalf);
      v8f acc = {};
#pragma unroll
      for (int kc = 0; kc < 8; ++kc) acc = wmma_f16(a[kc], w3b[kc], acc);
#pragma unroll
      for (int r = 0; r < 8; ++r) acc[r] += bv3;
      kout = acc;
    }
  };

  const int nsteps = num_steps[0];
  v8f k1 = {}, k2 = {}, k3 = {}, k4 = {}, k5 = {}, k6 = {};

#pragma unroll 1
  for (int step = 0; step < nsteps; ++step) {
    if (owner) stage_in(y);
    evalF(k1);
    if (owner) { v8f ys = y; axpy(ys, DT_H * CA21, k1); stage_in(ys); }
    evalF(k2);
    if (owner) { v8f ys = y; axpy(ys, DT_H * CA31, k1); axpy(ys, DT_H * CA32, k2); stage_in(ys); }
    evalF(k3);
    if (owner) {
      v8f ys = y;
      axpy(ys, DT_H * CA41, k1); axpy(ys, DT_H * CA42, k2); axpy(ys, DT_H * CA43, k3);
      stage_in(ys);
    }
    evalF(k4);
    if (owner) {
      v8f ys = y;
      axpy(ys, DT_H * CA51, k1); axpy(ys, DT_H * CA52, k2);
      axpy(ys, DT_H * CA53, k3); axpy(ys, DT_H * CA54, k4);
      stage_in(ys);
    }
    evalF(k5);
    if (owner) {
      v8f ys = y;
      axpy(ys, DT_H * CA61, k1); axpy(ys, DT_H * CA62, k2); axpy(ys, DT_H * CA63, k3);
      axpy(ys, DT_H * CA64, k4); axpy(ys, DT_H * CA65, k5);
      stage_in(ys);
    }
    evalF(k6);
    if (owner) {
      axpy(y, DT_H * CB1, k1); axpy(y, DT_H * CB2, k2); axpy(y, DT_H * CB3, k3);
      axpy(y, DT_H * CB4, k4); axpy(y, DT_H * CB5, k5); axpy(y, DT_H * CB6, k6);
    }
  }

  if (owner) {
    const int col = wave * 16 + lmod;
#pragma unroll
    for (int r = 0; r < 8; ++r)
      out[(rowBase + 8 * lhalf + r) * DIMD + col] = y[r];
  }
}

extern "C" void kernel_launch(void* const* d_in, const int* in_sizes, int n_in,
                              void* d_out, int out_size, void* d_ws, size_t ws_size,
                              hipStream_t stream) {
  (void)n_in; (void)out_size; (void)d_ws; (void)ws_size;
  const float* x0 = (const float*)d_in[0];
  const float* u  = (const float*)d_in[1];
  const float* W1 = (const float*)d_in[2];
  const float* b1 = (const float*)d_in[3];
  const float* W2 = (const float*)d_in[4];
  const float* b2 = (const float*)d_in[5];
  const float* W3 = (const float*)d_in[6];
  const float* b3 = (const float*)d_in[7];
  const int* nsteps = (const int*)d_in[8];
  float* out = (float*)d_out;

  const int B = in_sizes[0] / DIMD;   // 4096
  const int grid = B / 16;            // 256 workgroups, one 16-row tile each

  hipLaunchKernelGGL(tsit5_neural_ode_kernel, dim3(grid), dim3(256),
                     LDS_BYTES, stream,
                     x0, u, W1, b1, W2, b2, W3, b3, out, nsteps);
}
